// ProjectionSpaceRouter_53206054863471
// MI455X (gfx1250) — matrix operand order repaired and checked by reading, not compiled, not run on hardware
//
#include <hip/hip_runtime.h>
#include <cstdint>
#include <cstddef>

typedef unsigned short u16;
typedef __attribute__((ext_vector_type(16))) __bf16 v16bf;
typedef __attribute__((ext_vector_type(2)))  __bf16 v2bf;
typedef __attribute__((ext_vector_type(8)))  float  v8f;

union FragAB { v16bf v; uint4 q[2]; };

#define NTOK  32768
#define HDIM  1024
#define KTOT  3072
#define NFUSE 1536   // 1024 (space-gate hidden) + 512 (deviation-gate hidden)
#define NSP   64

// async-to-LDS builtin takes int4-vector pointers: AS1 (global) src, AS3 (local) dst
typedef int v4i __attribute__((vector_size(16)));
typedef __attribute__((address_space(1))) v4i as1_v4i;
typedef __attribute__((address_space(3))) v4i as3_v4i;

#if __has_builtin(__builtin_amdgcn_global_load_async_to_lds_b128)
#define HAVE_ASYNC_LDS 1
#else
#define HAVE_ASYNC_LDS 0
#endif

__device__ __forceinline__ void wait_asynccnt0() {
#if __has_builtin(__builtin_amdgcn_s_wait_asynccnt)
  __builtin_amdgcn_s_wait_asynccnt(0);
#else
  asm volatile("s_wait_asynccnt 0x0" ::: "memory");
#endif
}

// hardware RNE convert (v_cvt_bf16_f32)
__device__ __forceinline__ u16 f2bf(float f) {
  return __builtin_bit_cast(u16, (__bf16)f);
}
// pack two floats -> one dword of 2x bf16 with a single v_cvt_pk_bf16_f32
__device__ __forceinline__ unsigned pk2(float a, float b) {
#if __has_builtin(__builtin_amdgcn_cvt_pk_bf16_f32)
  auto r = __builtin_amdgcn_cvt_pk_bf16_f32(a, b);
  return __builtin_bit_cast(unsigned, r);
#else
  v2bf t;
  t.x = (__bf16)a;
  t.y = (__bf16)b;
  return __builtin_bit_cast(unsigned, t);
#endif
}
__device__ __forceinline__ float bf2f(u16 h) {
  return __uint_as_float(((unsigned)h) << 16);
}
__device__ __forceinline__ float gelu_exact(float x) {
  return 0.5f * x * (1.0f + erff(x * 0.70710678118654752f));
}

// ---------------- prepass: pack W1(1024x3072)+Wd1(512x3072) -> bf16 Wb(1536x3072), fused bias
__global__ __launch_bounds__(256) void pack_w_fused(const float* __restrict__ W1,
                                                    const float* __restrict__ Wd1,
                                                    const float* __restrict__ b1,
                                                    const float* __restrict__ bd1,
                                                    u16* __restrict__ Wb,
                                                    float* __restrict__ biasF) {
  const int row = blockIdx.x;
  const float* src = (row < 1024) ? (W1 + (size_t)row * KTOT)
                                  : (Wd1 + (size_t)(row - 1024) * KTOT);
  u16* dst = Wb + (size_t)row * KTOT;
  for (int v = threadIdx.x; v < KTOT / 4; v += 256) {
    float4 f = reinterpret_cast<const float4*>(src)[v];
    uint2 o;
    o.x = pk2(f.x, f.y);
    o.y = pk2(f.z, f.w);
    reinterpret_cast<uint2*>(dst)[v] = o;
  }
  if (threadIdx.x == 0)
    biasF[row] = (row < 1024) ? b1[row] : bd1[row - 1024];
}

// ---------------- prepass: pack W2(64x1024) -> bf16
__global__ __launch_bounds__(256) void pack_w2(const float* __restrict__ W2,
                                               u16* __restrict__ W2b) {
  const int row = blockIdx.x;
  const float* src = W2 + (size_t)row * HDIM;
  u16* dst = W2b + (size_t)row * HDIM;
  for (int v = threadIdx.x; v < HDIM / 4; v += 256) {
    float4 f = reinterpret_cast<const float4*>(src)[v];
    uint2 o;
    o.x = pk2(f.x, f.y);
    o.y = pk2(f.z, f.w);
    reinterpret_cast<uint2*>(dst)[v] = o;
  }
}

// ---------------- main fused GEMM: h = GELU([sm|tm|dg] @ Wb^T), bf16 out [NTOK, 1536]
// block = 256 thr (8 waves); tile 128(M) x 128(N); wave tile 32x64 (2x4 WMMA C frags)
// A (fp32 x) staged via VGPRs with v_cvt_pk_bf16_f32 convert (software-pipelined);
// B (bf16 Wb) staged via gfx1250 async-to-LDS DMA; epilogue restaged via LDS for
// coalesced 128B-per-thread global writes.
__global__ __launch_bounds__(256) void fused_hidden_gemm(
    const float* __restrict__ x0, const float* __restrict__ x1,
    const float* __restrict__ x2, const u16* __restrict__ Wb,
    const float* __restrict__ biasF, u16* __restrict__ hout) {
  // smem: sA = 128x32 bf16 (pitch 40), sB = 128x32 bf16 (pitch 40); reused as
  // 128x128 bf16 output tile (pitch 136) in the epilogue.
  __shared__ __align__(16) u16 smem[17408];
  u16* sA = smem;              // 5120 u16
  u16* sB = smem + 5120;       // 5120 u16

  const int tid  = threadIdx.x;
  const int lane = tid & 31;
  const int wid  = tid >> 5;
  const int waveM = wid & 3;     // 4 waves along M
  const int waveN = wid >> 2;    // 2 waves along N
  const int m0 = blockIdx.y * 128;
  const int n0 = blockIdx.x * 128;
  const int srow = tid >> 1;            // A staging: 2 threads per row
  const int sch  = (tid & 1) * 16;      // col half 0 / 16
  const int lhi = lane >> 4;            // ISA A/B fragment lane split
  const int llo = lane & 15;
  const int brow0 = tid >> 2;           // B staging rows (this thread): brow0, brow0+64
  const int bch   = (tid & 3) * 8;      // 16B chunk within the 64B row

  v8f acc[2][4];
  const v8f vz = {0.f,0.f,0.f,0.f,0.f,0.f,0.f,0.f};
  for (int i = 0; i < 2; i++) for (int j = 0; j < 4; j++) acc[i][j] = vz;

  // prologue: stage A regs for kk = 0
  float4 fA, fB, fC, fD;
  {
    const float4* g = reinterpret_cast<const float4*>(
        x0 + (size_t)(m0 + srow) * HDIM + sch);
    fA = g[0]; fB = g[1]; fC = g[2]; fD = g[3];
  }

  for (int kk = 0; kk < KTOT; kk += 32) {
    // convert staged fp32 -> packed bf16 (8 x v_cvt_pk_bf16_f32)
    uint4 p0, p1;
    p0.x = pk2(fA.x, fA.y);
    p0.y = pk2(fA.z, fA.w);
    p0.z = pk2(fB.x, fB.y);
    p0.w = pk2(fB.z, fB.w);
    p1.x = pk2(fC.x, fC.y);
    p1.y = pk2(fC.z, fC.w);
    p1.z = pk2(fD.x, fD.y);
    p1.w = pk2(fD.z, fD.w);

    __syncthreads();   // previous iteration done reading sA/sB
    *reinterpret_cast<uint4*>(&sA[srow * 40 + sch])     = p0;
    *reinterpret_cast<uint4*>(&sA[srow * 40 + sch + 8]) = p1;

    // B tile 128 rows x 64B: async DMA global -> LDS (2 x b128 per thread)
#if HAVE_ASYNC_LDS
    __builtin_amdgcn_global_load_async_to_lds_b128(
        (as1_v4i*)(uintptr_t)(Wb + (size_t)(n0 + brow0) * KTOT + kk + bch),
        (as3_v4i*)&sB[brow0 * 40 + bch], 0, 0);
    __builtin_amdgcn_global_load_async_to_lds_b128(
        (as1_v4i*)(uintptr_t)(Wb + (size_t)(n0 + brow0 + 64) * KTOT + kk + bch),
        (as3_v4i*)&sB[(brow0 + 64) * 40 + bch], 0, 0);
    wait_asynccnt0();
#else
    {
      uint4 t0 = *reinterpret_cast<const uint4*>(Wb + (size_t)(n0 + brow0) * KTOT + kk + bch);
      uint4 t1 = *reinterpret_cast<const uint4*>(Wb + (size_t)(n0 + brow0 + 64) * KTOT + kk + bch);
      *reinterpret_cast<uint4*>(&sB[brow0 * 40 + bch]) = t0;
      *reinterpret_cast<uint4*>(&sB[(brow0 + 64) * 40 + bch]) = t1;
    }
#endif
    __syncthreads();   // tiles visible

    // software pipeline: issue next A staging loads now (latency hides under WMMA)
    if (kk + 32 < KTOT) {
      const int kn = kk + 32;
      const int t = kn >> 10;
      const float* xb = (t == 0) ? x0 : ((t == 1) ? x1 : x2);
      const float4* g = reinterpret_cast<const float4*>(
          xb + (size_t)(m0 + srow) * HDIM + (kn & 1023) + sch);
      fA = g[0]; fB = g[1]; fC = g[2]; fD = g[3];
    }

    // A fragments (ISA 16-bit 16x32 layout: regs0-3 K=kb..kb+7, regs4-7 K=kb+16..kb+23)
    FragAB fa[2];
#pragma unroll
    for (int mt = 0; mt < 2; mt++) {
      const int m = waveM * 32 + mt * 16 + llo;
      const int kb = lhi * 8;
      fa[mt].q[0] = *reinterpret_cast<const uint4*>(&sA[m * 40 + kb]);
      fa[mt].q[1] = *reinterpret_cast<const uint4*>(&sA[m * 40 + kb + 16]);
    }
    // B fragments from LDS (K=lhi*8+0..7, lhi*8+16..23 at fixed N, same order as A)
    FragAB fb[4];
#pragma unroll
    for (int nt = 0; nt < 4; nt++) {
      const int nrow = waveN * 64 + nt * 16 + llo;
      fb[nt].q[0] = *reinterpret_cast<const uint4*>(&sB[nrow * 40 + lhi * 8]);
      fb[nt].q[1] = *reinterpret_cast<const uint4*>(&sB[nrow * 40 + lhi * 8 + 16]);
    }
#pragma unroll
    for (int nt = 0; nt < 4; nt++)
#pragma unroll
      for (int mt = 0; mt < 2; mt++)
        acc[mt][nt] = __builtin_amdgcn_wmma_f32_16x16x32_bf16(
            false, fa[mt].v, false, fb[nt].v, (short)0, acc[mt][nt], false, false);
  }

  // ---- epilogue: GELU -> bf16 tile in LDS (frag-by-frag so accs die early),
  //      then coalesced 128B-per-thread global stores.
  __syncthreads();
  u16* sOut = smem;   // 128 x 136 u16
#pragma unroll
  for (int mt = 0; mt < 2; mt++) {
#pragma unroll
    for (int nt = 0; nt < 4; nt++) {
      const int col_l = waveN * 64 + nt * 16 + llo;
      const float bias = biasF[n0 + col_l];
#pragma unroll
      for (int r = 0; r < 8; r++) {
        const int row_l = waveM * 32 + mt * 16 + r + lhi * 8;
        sOut[row_l * 136 + col_l] = f2bf(gelu_exact(acc[mt][nt][r] + bias));
      }
    }
  }
  __syncthreads();
  const int orow  = tid >> 1;
  const int ohalf = (tid & 1) * 64;
  const uint4* srcq = reinterpret_cast<const uint4*>(&sOut[orow * 136 + ohalf]);
  uint4* dstq = reinterpret_cast<uint4*>(hout + (size_t)(m0 + orow) * NFUSE + n0 + ohalf);
#pragma unroll
  for (int i = 0; i < 8; i++) dstq[i] = srcq[i];
}

// ---------------- logits (h[:, :1024] @ W2^T) + top-2 + softmax -> probs [NTOK, 64]
__global__ __launch_bounds__(256) void logits_topk_softmax(
    const u16* __restrict__ h, const u16* __restrict__ W2b,
    const float* __restrict__ b2, float* __restrict__ probs) {
  __shared__ float sl[128][65];   // padded: bank-conflict-free column scans
  const int tid  = threadIdx.x;
  const int lane = tid & 31;
  const int wid  = tid >> 5;
  const int lhi = lane >> 4;
  const int llo = lane & 15;
  const int m0 = blockIdx.x * 128;
  const int mrow = m0 + wid * 16 + llo;

  v8f acc[4];
  const v8f vz = {0.f,0.f,0.f,0.f,0.f,0.f,0.f,0.f};
  for (int i = 0; i < 4; i++) acc[i] = vz;

  const u16* arow = h + (size_t)mrow * NFUSE;   // first 1024 cols = space-gate hidden
  for (int k = 0; k < HDIM; k += 32) {
    FragAB fa;
    fa.q[0] = *reinterpret_cast<const uint4*>(arow + k + lhi * 8);
    fa.q[1] = *reinterpret_cast<const uint4*>(arow + k + lhi * 8 + 16);
#pragma unroll
    for (int nt = 0; nt < 4; nt++) {
      const int n = nt * 16 + llo;
      const u16* wrow = W2b + (size_t)n * HDIM + k + lhi * 8;
      FragAB fb;
      fb.q[0] = *reinterpret_cast<const uint4*>(wrow);
      fb.q[1] = *reinterpret_cast<const uint4*>(wrow + 16);
      acc[nt] = __builtin_amdgcn_wmma_f32_16x16x32_bf16(
          false, fa.v, false, fb.v, (short)0, acc[nt], false, false);
    }
  }
#pragma unroll
  for (int nt = 0; nt < 4; nt++) {
    const int n = nt * 16 + llo;
    const float bias = b2[n];
#pragma unroll
    for (int r = 0; r < 8; r++)
      sl[wid * 16 + r + lhi * 8][n] = acc[nt][r] + bias;
  }
  __syncthreads();

  if (tid < 128) {
    float v0 = -3.402823466e38f, v1 = -3.402823466e38f;
    int i0 = 0, i1 = 0;
    for (int j = 0; j < NSP; j++) {   // first-occurrence tie-break, matches lax.top_k
      const float v = sl[tid][j];
      if (v > v0)      { v1 = v0; i1 = i0; v0 = v; i0 = j; }
      else if (v > v1) { v1 = v;  i1 = j; }
    }
    // softmax over {v0, v1, 62 x -1e9}: exp(-1e9 - v0) underflows to exactly 0 in fp32
    const float e1 = expf(v1 - v0);
    const float inv = 1.0f / (1.0f + e1);
    const float p0 = inv, p1 = e1 * inv;
    float* orow = probs + (size_t)(m0 + tid) * NSP;
    for (int j = 0; j < NSP; j++)
      orow[j] = (j == i0) ? p0 : ((j == i1) ? p1 : 0.0f);
  }
}

// ---------------- beta = sigmoid(h[:, 1024:1536] @ Wd2 + bd2); one wave per token
__global__ __launch_bounds__(256) void beta_gemv(
    const u16* __restrict__ h, const float* __restrict__ Wd2,
    const float* __restrict__ bd2, float* __restrict__ beta) {
  const int lane = threadIdx.x & 31;
  const int wid  = threadIdx.x >> 5;
  const int row  = blockIdx.x * 8 + wid;
  const u16* hr = h + (size_t)row * NFUSE + 1024 + lane * 16;
  union { uint4 q; u16 e[8]; } a0, a1;
  a0.q = *reinterpret_cast<const uint4*>(hr);
  a1.q = *reinterpret_cast<const uint4*>(hr + 8);
  const float4* wv = reinterpret_cast<const float4*>(Wd2 + lane * 16);
  const float4 w0 = wv[0], w1 = wv[1], w2 = wv[2], w3 = wv[3];
  float dot = 0.f;
  dot += bf2f(a0.e[0])*w0.x + bf2f(a0.e[1])*w0.y + bf2f(a0.e[2])*w0.z + bf2f(a0.e[3])*w0.w;
  dot += bf2f(a0.e[4])*w1.x + bf2f(a0.e[5])*w1.y + bf2f(a0.e[6])*w1.z + bf2f(a0.e[7])*w1.w;
  dot += bf2f(a1.e[0])*w2.x + bf2f(a1.e[1])*w2.y + bf2f(a1.e[2])*w2.z + bf2f(a1.e[3])*w2.w;
  dot += bf2f(a1.e[4])*w3.x + bf2f(a1.e[5])*w3.y + bf2f(a1.e[6])*w3.z + bf2f(a1.e[7])*w3.w;
#pragma unroll
  for (int off = 16; off > 0; off >>= 1) dot += __shfl_xor(dot, off, 32);
  if (lane == 0) {
    const float z = dot + bd2[0];
    beta[row] = 1.0f / (1.0f + expf(-z));
  }
}

extern "C" void kernel_launch(void* const* d_in, const int* in_sizes, int n_in,
                              void* d_out, int out_size, void* d_ws, size_t ws_size,
                              hipStream_t stream) {
  const float* sm  = (const float*)d_in[0];
  const float* tm  = (const float*)d_in[1];
  const float* dg  = (const float*)d_in[2];
  const float* W1  = (const float*)d_in[3];
  const float* b1  = (const float*)d_in[4];
  const float* W2  = (const float*)d_in[5];
  const float* b2  = (const float*)d_in[6];
  const float* Wd1 = (const float*)d_in[7];
  const float* bd1 = (const float*)d_in[8];
  const float* Wd2 = (const float*)d_in[9];
  const float* bd2 = (const float*)d_in[10];

  // workspace layout (all offsets 256B-aligned):
  //   Wb    : u16[1536*3072]  @ 0          (9,437,184 B)
  //   W2b   : u16[64*1024]    @ 18,874,368 (   131,072 B)
  //   biasF : f32[1536]       @ 19,005,440 (     6,144 B)
  //   h     : u16[32768*1536] @ 19,011,584 (100,663,296 B)
  u16*   Wb    = (u16*)((char*)d_ws + 0);
  u16*   W2b   = (u16*)((char*)d_ws + 18874368);
  float* biasF = (float*)((char*)d_ws + 19005440);
  u16*   hbuf  = (u16*)((char*)d_ws + 19011584);

  float* probs = (float*)d_out;                       // [32768, 64]
  float* beta  = probs + (size_t)NTOK * NSP;          // [32768]

  pack_w_fused<<<NFUSE, 256, 0, stream>>>(W1, Wd1, b1, bd1, Wb, biasF);
  pack_w2<<<NSP, 256, 0, stream>>>(W2, W2b);
  fused_hidden_gemm<<<dim3(NFUSE / 128, NTOK / 128), 256, 0, stream>>>(
      sm, tm, dg, Wb, biasF, hbuf);
  logits_topk_softmax<<<NTOK / 128, 256, 0, stream>>>(hbuf, W2b, b2, probs);
  beta_gemv<<<NTOK / 8, 256, 0, stream>>>(hbuf, Wd2, bd2, beta);
}
